// TransformerSquared_16784732192936
// MI455X (gfx1250) — compile-verified, matrix operand order checked by reading, not run on hardware
//
#include <hip/hip_runtime.h>
#include <hip/hip_bf16.h>
#include <math.h>

// ---------------------------------------------------------------------------
// Types for CDNA5 WMMA (wave32): 16x16x32 bf16 -> f32
// ---------------------------------------------------------------------------
typedef __attribute__((ext_vector_type(16))) __bf16       v16bf;
typedef __attribute__((ext_vector_type(8)))  unsigned int v8u;
typedef __attribute__((ext_vector_type(8)))  float        v8f;

// pack two fp32 into one dword of bf16 (round-half-up via +0x8000, then one
// v_perm_b32 grabs both high halves: D = {hi16(b), hi16(a)})
__device__ __forceinline__ unsigned int pk2(float a, float b) {
  unsigned int ua = __builtin_bit_cast(unsigned int, a) + 0x8000u;
  unsigned int ub = __builtin_bit_cast(unsigned int, b) + 0x8000u;
  return __builtin_amdgcn_perm(ub, ua, 0x07060302u);
}

__device__ __forceinline__ uint4 pk8(float4 lo, float4 hi) {
  return make_uint4(pk2(lo.x, lo.y), pk2(lo.z, lo.w),
                    pk2(hi.x, hi.y), pk2(hi.z, hi.w));
}

__device__ __forceinline__ uint4 zsel(uint4 p, bool ok) {
  p.x = ok ? p.x : 0u;
  p.y = ok ? p.y : 0u;
  p.z = ok ? p.z : 0u;
  p.w = ok ? p.w : 0u;
  return p;
}

// ---------------------------------------------------------------------------
// Generic batched NT GEMM: C[M,N] (+)= A[M,K] * B[N,K]^T  (fp32 in/out,
// bf16 WMMA 16x16x32, f32 accumulate).
// Block tile 128(M) x 64(N) x 32(K), 256 threads = 8 waves in a 4(M) x 2(N)
// grid; each wave owns a 32x32 region = 2x2 WMMA tiles (4 wmma / K-step).
// Call-site contract (all satisfied here): A/B 16B-aligned, lda/ldb % 4 == 0,
// K % 16 == 0.
// ---------------------------------------------------------------------------
__global__ __launch_bounds__(256)
void gemm_nt_kernel(const float* __restrict__ A, const float* __restrict__ B,
                    float* __restrict__ C,
                    int M, int N, int K, int lda, int ldb, int ldcc,
                    long long sA, long long sB, long long sC, int accum) {
  __shared__ unsigned short As[128][40];   // 128 x 32 bf16 (+pad, 80B row)
  __shared__ unsigned short Bs[64][40];    //  64 x 32 bf16

  const float* Ab = A + (long long)blockIdx.z * sA;
  const float* Bb = B + (long long)blockIdx.z * sB;
  float*       Cb = C + (long long)blockIdx.z * sC;

  const int n0   = blockIdx.x * 64;
  const int m0   = blockIdx.y * 128;
  const int tid  = threadIdx.x;
  const int lane = tid & 31;
  const int wave = tid >> 5;
  const int wn   = wave & 1;        // N sub-block (32 cols)
  const int wm   = wave >> 1;       // M sub-block (32 rows), 0..3
  const int row  = lane & 15;
  const int grp  = lane >> 4;

  // staging assignments
  const int ar = tid >> 1;          // 0..127, A row
  const int ak = (tid & 1) << 4;    // 0 or 16 (16 floats per thread)
  const int br = tid >> 2;          // 0..63,  B row
  const int bk = (tid & 3) << 3;    // 0,8,16,24 (8 floats per thread)

  const bool okAm = (m0 + ar) < M;
  const bool okBn = (n0 + br) < N;
  const long long arow = (long long)(m0 + ar) * lda;
  const long long brow = (long long)(n0 + br) * ldb;

  v8f acc00 = {}, acc01 = {}, acc10 = {}, acc11 = {};

  for (int k0 = 0; k0 < K; k0 += 32) {
    // ---- stage fp32 -> bf16 tiles into LDS ----
    // K % 16 == 0 and chunk starts are 16/8-aligned => a thread's chunk is
    // valid all-or-nothing: one predicate, one clamped base, vector loads.
    {
      const bool okA = okAm && (k0 + ak < K);
      const float4* pa = (const float4*)(Ab + (okA ? (arow + k0 + ak) : 0));
      float4 a0 = pa[0], a1 = pa[1], a2 = pa[2], a3 = pa[3];
      uint4 p0 = zsel(pk8(a0, a1), okA);
      uint4 p1 = zsel(pk8(a2, a3), okA);
      *(uint4*)&As[ar][ak]     = p0;
      *(uint4*)&As[ar][ak + 8] = p1;

      const bool okB = okBn && (k0 + bk < K);
      const float4* pb = (const float4*)(Bb + (okB ? (brow + k0 + bk) : 0));
      float4 b0 = pb[0], b1 = pb[1];
      uint4 q0 = zsel(pk8(b0, b1), okB);
      *(uint4*)&Bs[br][bk] = q0;

      // speculative prefetch of next K tile (global_prefetch_b8)
      if (k0 + 32 < K) {
        if (okAm) __builtin_prefetch(Ab + arow + k0 + 32 + ak, 0, 1);
        if (okBn) __builtin_prefetch(Bb + brow + k0 + 32 + bk, 0, 1);
      }
    }
    __syncthreads();

    // ---- fragments per ISA 16-bit layout: lane (row=L%16, grp=L/16);
    //      element e<8 -> k = e + grp*8 ; e>=8 -> k = e + 8 + grp*8
    const int ka = grp * 8;
    uint4 al0 = *(const uint4*)&As[wm * 32 + row][ka];
    uint4 ah0 = *(const uint4*)&As[wm * 32 + row][ka + 16];
    uint4 al1 = *(const uint4*)&As[wm * 32 + 16 + row][ka];
    uint4 ah1 = *(const uint4*)&As[wm * 32 + 16 + row][ka + 16];
    uint4 bl0 = *(const uint4*)&Bs[wn * 32 + row][ka];
    uint4 bh0 = *(const uint4*)&Bs[wn * 32 + row][ka + 16];
    uint4 bl1 = *(const uint4*)&Bs[wn * 32 + 16 + row][ka];
    uint4 bh1 = *(const uint4*)&Bs[wn * 32 + 16 + row][ka + 16];

    v8u u;
    u[0]=al0.x; u[1]=al0.y; u[2]=al0.z; u[3]=al0.w;
    u[4]=ah0.x; u[5]=ah0.y; u[6]=ah0.z; u[7]=ah0.w;
    v16bf fa0 = __builtin_bit_cast(v16bf, u);
    u[0]=al1.x; u[1]=al1.y; u[2]=al1.z; u[3]=al1.w;
    u[4]=ah1.x; u[5]=ah1.y; u[6]=ah1.z; u[7]=ah1.w;
    v16bf fa1 = __builtin_bit_cast(v16bf, u);
    u[0]=bl0.x; u[1]=bl0.y; u[2]=bl0.z; u[3]=bl0.w;
    u[4]=bh0.x; u[5]=bh0.y; u[6]=bh0.z; u[7]=bh0.w;
    v16bf fb0 = __builtin_bit_cast(v16bf, u);
    u[0]=bl1.x; u[1]=bl1.y; u[2]=bl1.z; u[3]=bl1.w;
    u[4]=bh1.x; u[5]=bh1.y; u[6]=bh1.z; u[7]=bh1.w;
    v16bf fb1 = __builtin_bit_cast(v16bf, u);

    acc00 = __builtin_amdgcn_wmma_f32_16x16x32_bf16(false, fa0, false, fb0,
                                                    (short)0, acc00, false, false);
    acc10 = __builtin_amdgcn_wmma_f32_16x16x32_bf16(false, fa1, false, fb0,
                                                    (short)0, acc10, false, false);
    acc01 = __builtin_amdgcn_wmma_f32_16x16x32_bf16(false, fa0, false, fb1,
                                                    (short)0, acc01, false, false);
    acc11 = __builtin_amdgcn_wmma_f32_16x16x32_bf16(false, fa1, false, fb1,
                                                    (short)0, acc11, false, false);
    __syncthreads();
  }

  // ---- epilogue: C/D layout -> lane column = lane%16, VGPR j -> row j+grp*8
  const int gnb = n0 + wn * 32 + row;
  const int gmb = m0 + wm * 32 + grp * 8;
#pragma unroll
  for (int j = 0; j < 8; ++j) {
    const int gm0 = gmb + j, gm1 = gmb + 16 + j;
    if (gnb < N) {
      if (gm0 < M) {
        long long o = (long long)gm0 * ldcc + gnb;
        Cb[o] = accum ? (Cb[o] + acc00[j]) : acc00[j];
      }
      if (gm1 < M) {
        long long o = (long long)gm1 * ldcc + gnb;
        Cb[o] = accum ? (Cb[o] + acc10[j]) : acc10[j];
      }
    }
    if (gnb + 16 < N) {
      if (gm0 < M) {
        long long o = (long long)gm0 * ldcc + gnb + 16;
        Cb[o] = accum ? (Cb[o] + acc01[j]) : acc01[j];
      }
      if (gm1 < M) {
        long long o = (long long)gm1 * ldcc + gnb + 16;
        Cb[o] = accum ? (Cb[o] + acc11[j]) : acc11[j];
      }
    }
  }
}

// ---------------------------------------------------------------------------
// Elementwise / reduction kernels (fp32)
// ---------------------------------------------------------------------------
__global__ void embed_kernel(const int* __restrict__ idx,
                             const float* __restrict__ tok,
                             const float* __restrict__ pos,
                             float* __restrict__ x, int T, int D) {
  long long i = (long long)blockIdx.x * 256 + threadIdx.x;
  if (i < (long long)T * D) {
    int t = (int)(i / D), d = (int)(i % D);
    x[i] = tok[(long long)idx[t] * D + d] + pos[i];
  }
}

__global__ void layernorm_kernel(const float* __restrict__ x,
                                 const float* __restrict__ w,
                                 const float* __restrict__ b,
                                 float* __restrict__ out, int D) {
  __shared__ float red[256];
  const long long r = blockIdx.x;
  const float* xr = x + r * D;
  float* outr = out + r * D;
  const int tid = threadIdx.x;

  float s = 0.0f;
  for (int j = tid; j < D; j += 256) s += xr[j];
  red[tid] = s; __syncthreads();
  for (int st = 128; st > 0; st >>= 1) {
    if (tid < st) red[tid] += red[tid + st];
    __syncthreads();
  }
  const float mean = red[0] / (float)D;
  __syncthreads();

  float v = 0.0f;
  for (int j = tid; j < D; j += 256) { float d = xr[j] - mean; v += d * d; }
  red[tid] = v; __syncthreads();
  for (int st = 128; st > 0; st >>= 1) {
    if (tid < st) red[tid] += red[tid + st];
    __syncthreads();
  }
  const float rstd = rsqrtf(red[0] / (float)D + 1e-5f);
  __syncthreads();

  for (int j = tid; j < D; j += 256)
    outr[j] = (xr[j] - mean) * rstd * w[j] + b[j];
}

__global__ void scale_cols_kernel(float* __restrict__ t,
                                  const float* __restrict__ z,
                                  int cols, long long n) {
  long long i = (long long)blockIdx.x * 256 + threadIdx.x;
  if (i < n) t[i] *= z[i % cols];
}

__global__ void gelu_kernel(float* __restrict__ t, long long n) {
  long long i = (long long)blockIdx.x * 256 + threadIdx.x;
  if (i < n) {
    float v = t[i];
    t[i] = 0.5f * v * (1.0f + erff(v * 0.70710678118654752f));
  }
}

// one block per row t; applies scale, causal mask (cols > t -> 0), softmax
__global__ void softmax_causal_kernel(float* __restrict__ w, int T, float scale) {
  __shared__ float red[256];
  const int t = blockIdx.x;
  float* row = w + (long long)t * T;
  const int tid = threadIdx.x;

  float m = -3.0e38f;
  for (int j = tid; j <= t; j += 256) m = fmaxf(m, row[j] * scale);
  red[tid] = m; __syncthreads();
  for (int st = 128; st > 0; st >>= 1) {
    if (tid < st) red[tid] = fmaxf(red[tid], red[tid + st]);
    __syncthreads();
  }
  m = red[0];
  __syncthreads();

  float s = 0.0f;
  for (int j = tid; j <= t; j += 256) s += expf(row[j] * scale - m);
  red[tid] = s; __syncthreads();
  for (int st = 128; st > 0; st >>= 1) {
    if (tid < st) red[tid] += red[tid + st];
    __syncthreads();
  }
  const float inv = 1.0f / red[0];
  __syncthreads();

  for (int j = tid; j < T; j += 256)
    row[j] = (j <= t) ? expf(row[j] * scale - m) * inv : 0.0f;
}

// ---------------------------------------------------------------------------
// Host orchestration
// ---------------------------------------------------------------------------
extern "C" void kernel_launch(void* const* d_in, const int* in_sizes, int n_in,
                              void* d_out, int out_size, void* d_ws, size_t ws_size,
                              hipStream_t stream) {
  const int*   idx     = (const int*)  d_in[0];
  const float* tok_emb = (const float*)d_in[1];
  const float* pos_emb = (const float*)d_in[2];
  const float* ln1_w   = (const float*)d_in[3];
  const float* ln1_b   = (const float*)d_in[4];
  const float* ln2_w   = (const float*)d_in[5];
  const float* ln2_b   = (const float*)d_in[6];
  const float* kqv_V   = (const float*)d_in[7];
  const float* kqv_z   = (const float*)d_in[8];
  const float* kqv_U   = (const float*)d_in[9];
  const float* proj_V  = (const float*)d_in[10];
  const float* proj_z  = (const float*)d_in[11];
  const float* proj_U  = (const float*)d_in[12];
  const float* f1_V    = (const float*)d_in[13];
  const float* f1_z    = (const float*)d_in[14];
  const float* f1_U    = (const float*)d_in[15];
  const float* f2_V    = (const float*)d_in[16];
  const float* f2_z    = (const float*)d_in[17];
  const float* f2_U    = (const float*)d_in[18];
  const float* lnf_w   = (const float*)d_in[19];
  const float* lnf_b   = (const float*)d_in[20];
  const float* lm_w    = (const float*)d_in[21];
  float* logits = (float*)d_out;

  const int T = 2048, D = 1024, H = 16, HS = 64, L = 4, FF = 4096;
  const int RH = 16, RP = 256, RF = 256, V = 32000;
  const int GHR = 3 * H * RH;                 // 768 flattened (g,h,r) columns

  // workspace layout (fp32), total ~= 99 MB
  float* ws   = (float*)d_ws;
  float* x    = ws;                                   // [T, D]
  float* hbuf = x    + (long long)T * D;              // [T, D]   (ln out / att concat)
  float* tbuf = hbuf + (long long)T * D;              // [T, 768] (SVF intermediates)
  float* qb   = tbuf + (long long)T * GHR;            // [H, T, HS]
  float* kb   = qb   + (long long)H * T * HS;         // [H, T, HS]
  float* vTb  = kb   + (long long)H * T * HS;         // [H, HS, T]  (v transposed)
  float* wei  = vTb  + (long long)H * HS * T;         // [T, T]  (per-head, reused)
  float* ffb  = wei  + (long long)T * T;              // [T, FF]

  auto gemm = [&](const float* A, const float* B, float* C, int M, int N, int K,
                  int lda, int ldb, int ldcc, long long sA, long long sB,
                  long long sC, int batch, int accum) {
    dim3 g((unsigned)((N + 63) / 64), (unsigned)((M + 127) / 128), (unsigned)batch);
    gemm_nt_kernel<<<g, 256, 0, stream>>>(A, B, C, M, N, K, lda, ldb, ldcc,
                                          sA, sB, sC, accum);
  };

  // x = tok_emb[idx] + pos_emb
  embed_kernel<<<(T * D + 255) / 256, 256, 0, stream>>>(idx, tok_emb, pos_emb, x, T, D);

  for (int l = 0; l < L; ++l) {
    // h = ln1(x)
    layernorm_kernel<<<T, 256, 0, stream>>>(x, ln1_w + l * D, ln1_b + l * D, hbuf, D);

    // t[t,(g,h,r)] = h @ kqv_V[l]^T, then * z
    gemm(hbuf, kqv_V + (long long)l * GHR * D, tbuf, T, GHR, D, D, D, GHR,
         0, 0, 0, 1, 0);
    scale_cols_kernel<<<(unsigned)(((long long)T * GHR + 255) / 256), 256, 0, stream>>>(
        tbuf, kqv_z + l * GHR, GHR, (long long)T * GHR);

    const long long Ul = (long long)l * 3 * H * HS * RH;  // kqv_U layer offset
    // k (g=0): [H][T,HS] = t_head @ U_head^T
    gemm(tbuf + 0 * H * RH, kqv_U + Ul + 0LL * H * HS * RH, kb,
         T, HS, RH, GHR, RH, HS, RH, (long long)HS * RH, (long long)T * HS, H, 0);
    // q (g=1)
    gemm(tbuf + 1 * H * RH, kqv_U + Ul + 1LL * H * HS * RH, qb,
         T, HS, RH, GHR, RH, HS, RH, (long long)HS * RH, (long long)T * HS, H, 0);
    // vT (g=2): [H][HS,T] = U_head @ t_head^T   (transposed so att@v is NT)
    gemm(kqv_U + Ul + 2LL * H * HS * RH, tbuf + 2 * H * RH, vTb,
         HS, T, RH, RH, GHR, T, (long long)HS * RH, RH, (long long)HS * T, H, 0);

    // attention per head (wei buffer reused; sequential on stream)
    for (int h = 0; h < H; ++h) {
      gemm(qb + (long long)h * T * HS, kb + (long long)h * T * HS, wei,
           T, T, HS, HS, HS, T, 0, 0, 0, 1, 0);
      softmax_causal_kernel<<<T, 256, 0, stream>>>(wei, T, 0.125f);  // HS^-0.5
      // att[:, h*HS : h*HS+HS] = wei @ v  (head-concat directly into hbuf)
      gemm(wei, vTb + (long long)h * HS * T, hbuf + h * HS,
           T, HS, T, T, T, D, 0, 0, 0, 1, 0);
    }

    // x += ((att @ proj_V^T) * z) @ proj_U^T
    gemm(hbuf, proj_V + (long long)l * RP * D, tbuf, T, RP, D, D, D, RP,
         0, 0, 0, 1, 0);
    scale_cols_kernel<<<(unsigned)(((long long)T * RP + 255) / 256), 256, 0, stream>>>(
        tbuf, proj_z + l * RP, RP, (long long)T * RP);
    gemm(tbuf, proj_U + (long long)l * D * RP, x, T, D, RP, RP, RP, D,
         0, 0, 0, 1, 1);

    // ffn: x += svf2(gelu(svf1(ln2(x))))
    layernorm_kernel<<<T, 256, 0, stream>>>(x, ln2_w + l * D, ln2_b + l * D, hbuf, D);
    gemm(hbuf, f1_V + (long long)l * RF * D, tbuf, T, RF, D, D, D, RF,
         0, 0, 0, 1, 0);
    scale_cols_kernel<<<(unsigned)(((long long)T * RF + 255) / 256), 256, 0, stream>>>(
        tbuf, f1_z + l * RF, RF, (long long)T * RF);
    gemm(tbuf, f1_U + (long long)l * FF * RF, ffb, T, FF, RF, RF, RF, FF,
         0, 0, 0, 1, 0);
    gelu_kernel<<<(unsigned)(((long long)T * FF + 255) / 256), 256, 0, stream>>>(
        ffb, (long long)T * FF);
    gemm(ffb, f2_V + (long long)l * RF * FF, tbuf, T, RF, FF, FF, FF, RF,
         0, 0, 0, 1, 0);
    scale_cols_kernel<<<(unsigned)(((long long)T * RF + 255) / 256), 256, 0, stream>>>(
        tbuf, f2_z + l * RF, RF, (long long)T * RF);
    gemm(tbuf, f2_U + (long long)l * D * RF, x, T, D, RF, RF, RF, D,
         0, 0, 0, 1, 1);
  }

  // logits = ln_f(x) @ lm_w^T
  layernorm_kernel<<<T, 256, 0, stream>>>(x, lnf_w, lnf_b, hbuf, D);
  gemm(hbuf, lm_w, logits, T, V, D, D, D, V, 0, 0, 0, 1, 0);
}
